// MultiHeadAttention_4131758539005
// MI455X (gfx1250) — matrix operand order repaired and checked by reading, not compiled, not run on hardware
//
#include <hip/hip_runtime.h>
#include <hip/hip_bf16.h>

// ---------------------------------------------------------------------------
// MHA forward for MI455X (gfx1250): f16 WMMA everywhere, f32 accumulation.
// B=4, S=2048, H=16, Dh=64, D=1024.
// ---------------------------------------------------------------------------

#define B_SZ   4
#define S_LEN  2048
#define NHEAD  16
#define HDIM   64
#define DMODEL 1024
#define NEG_MIN (-3.402823466e38f)

typedef _Float16 v16h __attribute__((ext_vector_type(16)));
typedef _Float16 v8h  __attribute__((ext_vector_type(8)));
typedef float    v8f  __attribute__((ext_vector_type(8)));
typedef float    v4f  __attribute__((ext_vector_type(4)));

__device__ __forceinline__ v8f wmma16(v16h a, v16h b, v8f c) {
    return __builtin_amdgcn_wmma_f32_16x16x32_f16(
        /*neg_a=*/false, a, /*neg_b=*/false, b,
        /*c_mod=*/(short)0, c, /*reuse_a=*/false, /*reuse_b=*/false);
}

__device__ __forceinline__ v8f zero8() {
    v8f z;
#pragma unroll
    for (int i = 0; i < 8; ++i) z[i] = 0.0f;
    return z;
}

// A fragment (16x32 f16). Lane row = lane&15; element e -> K = 16*(lane>=16)+e.
// p = row base + kk.
// Works for both global and LDS pointers.
__device__ __forceinline__ v16h load_a_f16(const _Float16* p, int hs) {
    v8h lo = *(const v8h*)(p + hs * 16);
    v8h hi = *(const v8h*)(p + hs * 16 + 8);
    v16h a;
#pragma unroll
    for (int e = 0; e < 8; ++e) { a[e] = lo[e]; a[8 + e] = hi[e]; }
    return a;
}

// B fragment (32x16 f16). Lanes 0-15 hold K=0..15, lanes 16-31 hold K=16..31,
// column = lane&15. 16 elements contiguous in K: p = col*ld + kk + hs*16.
__device__ __forceinline__ v16h load16h(const _Float16* __restrict__ p) {
    v8h lo = *(const v8h*)p;
    v8h hi = *(const v8h*)(p + 8);
    v16h a;
#pragma unroll
    for (int e = 0; e < 8; ++e) { a[e] = lo[e]; a[8 + e] = hi[e]; }
    return a;
}

// ---------------------------------------------------------------------------
// Weight transpose + f16 convert: Wt[n][k] = (f16) W[k][n]
// ---------------------------------------------------------------------------
__global__ __launch_bounds__(256) void transpose_w_kernel(
    const float* __restrict__ W, _Float16* __restrict__ Wt) {
    int idx = blockIdx.x * 256 + threadIdx.x;     // 0 .. 1024*1024-1
    int n = idx >> 10, k = idx & 1023;
    Wt[idx] = (_Float16)W[k * DMODEL + n];
}

// ---------------------------------------------------------------------------
// Projection GEMM: Out = X(f32, 8192x1024) @ W (via Wt f16 [N][K]), f16 out.
// Block = 8 waves sharing one 16-row M tile; each wave owns a 64-col N block.
// The block's A panel (16x1024) is staged to LDS as f16 ONCE (cooperative
// load + convert), so the hot loop is: 2 ds_load_b128 (A) + 8 global b128 (B)
// + 4 independent WMMAs per 32-K step. No conversions, no redundant A reads.
// mode 0: out[((b*H+h)*S + s)*64 + d]   (Q,K head-major)
// mode 1: out[((b*H+h)*64 + d)*S + s]   (V transposed for P.V B-fragments)
// ---------------------------------------------------------------------------
__global__ __launch_bounds__(256) void proj_kernel(
    const float* __restrict__ X, const _Float16* __restrict__ Wt,
    _Float16* __restrict__ Out, int mode) {
    __shared__ _Float16 apanel[16][DMODEL];       // 32 KB

    int wave = threadIdx.x >> 5, lane = threadIdx.x & 31;
    int tid  = threadIdx.x;
    int tm = blockIdx.x >> 1;                     // 16-row tile (shared by block)
    int tq = (blockIdx.x * 8 + wave) & 15;        // 64-col block per wave
    int hs = lane >> 4, l15 = lane & 15;

    // ---- cooperative stage: global f32 -> LDS f16 (convert once per block) ----
    {
        int r  = tid >> 4;                        // 0..15
        int c0 = (tid & 15) * 64;                 // 0..960
        const float* src = X + (size_t)(tm * 16 + r) * DMODEL + c0;
        _Float16* dst = &apanel[r][c0];
#pragma unroll
        for (int i = 0; i < 8; ++i) {
            v4f x0 = *(const v4f*)(src + i * 8);
            v4f x1 = *(const v4f*)(src + i * 8 + 4);
            v8h hp;
#pragma unroll
            for (int e = 0; e < 4; ++e) {
                hp[e]     = (_Float16)x0[e];
                hp[4 + e] = (_Float16)x1[e];
            }
            *(v8h*)(dst + i * 8) = hp;
        }
    }
    __syncthreads();

    const _Float16* bcol0 = Wt + (size_t)(tq * 64 + 0 * 16 + l15) * DMODEL;
    const _Float16* bcol1 = Wt + (size_t)(tq * 64 + 1 * 16 + l15) * DMODEL;
    const _Float16* bcol2 = Wt + (size_t)(tq * 64 + 2 * 16 + l15) * DMODEL;
    const _Float16* bcol3 = Wt + (size_t)(tq * 64 + 3 * 16 + l15) * DMODEL;

    v8f acc[4];
#pragma unroll
    for (int j = 0; j < 4; ++j) acc[j] = zero8();

#pragma unroll 2
    for (int kk = 0; kk < DMODEL; kk += 32) {
        // fetch everything first, then 4 back-to-back WMMAs
        v16h b0 = load16h(bcol0 + kk + hs * 16);
        v16h b1 = load16h(bcol1 + kk + hs * 16);
        v16h b2 = load16h(bcol2 + kk + hs * 16);
        v16h b3 = load16h(bcol3 + kk + hs * 16);
        v16h a  = load_a_f16(&apanel[l15][kk], hs);
        acc[0] = wmma16(a, b0, acc[0]);
        acc[1] = wmma16(a, b1, acc[1]);
        acc[2] = wmma16(a, b2, acc[2]);
        acc[3] = wmma16(a, b3, acc[3]);
    }

#pragma unroll
    for (int j = 0; j < 4; ++j) {
#pragma unroll
        for (int v = 0; v < 8; ++v) {
            int token = tm * 16 + v + hs * 8;     // C row
            int col   = tq * 64 + j * 16 + l15;   // C col (dmodel)
            int b = token >> 11, s = token & (S_LEN - 1);
            int h = col >> 6,    d = col & (HDIM - 1);
            size_t idx;
            if (mode == 0) idx = ((size_t)(b * NHEAD + h) * S_LEN + s) * HDIM + d;
            else           idx = ((size_t)(b * NHEAD + h) * HDIM + d) * S_LEN + s;
            Out[idx] = (_Float16)acc[j][v];
        }
    }
}

// ---------------------------------------------------------------------------
// Flash attention: one wave = one (b, h, 16-query tile). 32 keys / iteration.
// Head index is the FASTEST-varying wave coordinate so the 8 waves of a block
// are 8 heads of the same (b, q-tile): they issue identical mask addresses,
// turning the 16x mask re-read (~1 GB) into L0/L2 hits.
// qh, kh: f16 [b][h][s][64]; vt: f16 [b][h][64][s]; mask: f32 [b][sq][sk];
// aw out: f16 [b][s][1024].
// ---------------------------------------------------------------------------
__global__ __launch_bounds__(256) void attn_kernel(
    const _Float16* __restrict__ qh, const _Float16* __restrict__ kh,
    const _Float16* __restrict__ vt, const float* __restrict__ mask,
    _Float16* __restrict__ aw) {
    __shared__ _Float16 pbuf[8][16][32];          // per-wave P staging (8 KB)

    int wave = threadIdx.x >> 5, lane = threadIdx.x & 31;
    int hs = lane >> 4, l15 = lane & 15;
    int gid = blockIdx.x * 8 + wave;              // 0 .. 8191
    int h  = gid & (NHEAD - 1);                   // head fastest
    int qt = (gid >> 4) & 127;                    // S/16 tiles
    int b  = gid >> 11;
    int bh = b * NHEAD + h;

    const _Float16* qp = qh + (size_t)bh * S_LEN * HDIM;
    const _Float16* kp = kh + (size_t)bh * S_LEN * HDIM;
    const _Float16* vp = vt + (size_t)bh * HDIM * S_LEN;

    int q0 = qt * 16;
    // Q A-fragments for d=0..31 and d=32..63 (held for whole loop)
    v16h aq0 = load_a_f16(qp + (size_t)(q0 + l15) * HDIM + 0,  hs);
    v16h aq1 = load_a_f16(qp + (size_t)(q0 + l15) * HDIM + 32, hs);

    v8f o0 = zero8(), o1 = zero8(), o2 = zero8(), o3 = zero8();
    float m_i[8], l_i[8];
#pragma unroll
    for (int v = 0; v < 8; ++v) { m_i[v] = NEG_MIN; l_i[v] = 0.0f; }

    const float scale = 0.125f;                   // 1/sqrt(64)
    int kend = q0 + 16;                           // causal bound (exclusive)

    for (int kb = 0; kb < kend; kb += 32) {
        // ---- S = Q K^T for two 16-key subtiles (2 independent chains) ----
        v8f s0 = zero8(), s1 = zero8();
        {
            v16h bk00 = load16h(kp + (size_t)(kb + l15) * HDIM + 0 + hs * 16);
            v16h bk10 = load16h(kp + (size_t)(kb + 16 + l15) * HDIM + 0 + hs * 16);
            v16h bk01 = load16h(kp + (size_t)(kb + l15) * HDIM + 32 + hs * 16);
            v16h bk11 = load16h(kp + (size_t)(kb + 16 + l15) * HDIM + 32 + hs * 16);
            s0 = wmma16(aq0, bk00, s0);
            s1 = wmma16(aq0, bk10, s1);
            s0 = wmma16(aq1, bk01, s0);
            s1 = wmma16(aq1, bk11, s1);
        }

        // ---- mask + causal:  s = s*m + (1-m)*NEG_MIN ----
        float nmax[8];
#pragma unroll
        for (int v = 0; v < 8; ++v) {
            int qrow = q0 + v + hs * 8;
            int key0 = kb + l15;
            int key1 = kb + 16 + l15;
            const float* mrow = mask + (size_t)(b * S_LEN + qrow) * S_LEN;
            float m0 = (key0 <= qrow) ? mrow[key0] : 0.0f;
            float m1 = (key1 <= qrow) ? mrow[key1] : 0.0f;
            float sv0 = s0[v] * scale, sv1 = s1[v] * scale;
            sv0 = sv0 * m0 + (1.0f - m0) * NEG_MIN;
            sv1 = sv1 * m1 + (1.0f - m1) * NEG_MIN;
            s0[v] = sv0; s1[v] = sv1;
            nmax[v] = fmaxf(sv0, sv1);
        }
        // row max across the 16 lanes holding each row (halves stay disjoint)
#pragma unroll
        for (int v = 0; v < 8; ++v) {
            float t = nmax[v];
            t = fmaxf(t, __shfl_xor(t, 1, 32));
            t = fmaxf(t, __shfl_xor(t, 2, 32));
            t = fmaxf(t, __shfl_xor(t, 4, 32));
            t = fmaxf(t, __shfl_xor(t, 8, 32));
            nmax[v] = t;
        }
        // ---- online softmax update ----
        float alpha[8], p0[8], p1[8];
#pragma unroll
        for (int v = 0; v < 8; ++v) {
            float mn = fmaxf(m_i[v], nmax[v]);
            alpha[v] = __expf(m_i[v] - mn);
            m_i[v] = mn;
            p0[v] = __expf(s0[v] - mn);
            p1[v] = __expf(s1[v] - mn);
            float rs = p0[v] + p1[v];
            rs += __shfl_xor(rs, 1, 32);
            rs += __shfl_xor(rs, 2, 32);
            rs += __shfl_xor(rs, 4, 32);
            rs += __shfl_xor(rs, 8, 32);
            l_i[v] = l_i[v] * alpha[v] + rs;
        }
#pragma unroll
        for (int v = 0; v < 8; ++v) {
            o0[v] *= alpha[v]; o1[v] *= alpha[v];
            o2[v] *= alpha[v]; o3[v] *= alpha[v];
        }

        // ---- relayout P (C layout -> A fragment) through per-wave LDS ----
#pragma unroll
        for (int v = 0; v < 8; ++v) {
            int r = v + hs * 8;
            pbuf[wave][r][l15]      = (_Float16)p0[v];
            pbuf[wave][r][16 + l15] = (_Float16)p1[v];
        }
        // DS ops from one wave are in-order; wait for data + fence compiler.
        asm volatile("s_wait_dscnt 0" ::: "memory");
        v16h pa = load_a_f16(&pbuf[wave][l15][0], hs);

        // ---- O += P V  (4 N-tiles of 16 dims, contraction over 32 keys) ----
        v16h bv0 = load16h(vp + (size_t)(0 * 16 + l15) * S_LEN + kb + hs * 16);
        v16h bv1 = load16h(vp + (size_t)(1 * 16 + l15) * S_LEN + kb + hs * 16);
        v16h bv2 = load16h(vp + (size_t)(2 * 16 + l15) * S_LEN + kb + hs * 16);
        v16h bv3 = load16h(vp + (size_t)(3 * 16 + l15) * S_LEN + kb + hs * 16);
        o0 = wmma16(pa, bv0, o0);
        o1 = wmma16(pa, bv1, o1);
        o2 = wmma16(pa, bv2, o2);
        o3 = wmma16(pa, bv3, o3);
    }

    // ---- normalize and write to aw[b][q][h*64 + d] ----
#pragma unroll
    for (int v = 0; v < 8; ++v) {
        int qrow = q0 + v + hs * 8;
        float inv = 1.0f / l_i[v];
        _Float16* orow = aw + (size_t)(b * S_LEN + qrow) * DMODEL + h * HDIM;
        orow[0 * 16 + l15] = (_Float16)(o0[v] * inv);
        orow[1 * 16 + l15] = (_Float16)(o1[v] * inv);
        orow[2 * 16 + l15] = (_Float16)(o2[v] * inv);
        orow[3 * 16 + l15] = (_Float16)(o3[v] * inv);
    }
}

// ---------------------------------------------------------------------------
// Output projection: d_out(f32) = aw(f16 8192x1024) @ Wo (via Wt f16 [N][K]).
// Same block structure as proj_kernel; A panel staged to LDS (f16 copy).
// ---------------------------------------------------------------------------
__global__ __launch_bounds__(256) void outproj_kernel(
    const _Float16* __restrict__ A, const _Float16* __restrict__ Wt,
    float* __restrict__ Out) {
    __shared__ _Float16 apanel[16][DMODEL];       // 32 KB

    int wave = threadIdx.x >> 5, lane = threadIdx.x & 31;
    int tid  = threadIdx.x;
    int tm = blockIdx.x >> 1;
    int tq = (blockIdx.x * 8 + wave) & 15;
    int hs = lane >> 4, l15 = lane & 15;

    // ---- cooperative stage: global f16 -> LDS f16 ----
    {
        int r  = tid >> 4;
        int c0 = (tid & 15) * 64;
        const _Float16* src = A + (size_t)(tm * 16 + r) * DMODEL + c0;
        _Float16* dst = &apanel[r][c0];
#pragma unroll
        for (int i = 0; i < 8; ++i)
            *(v8h*)(dst + i * 8) = *(const v8h*)(src + i * 8);
    }
    __syncthreads();

    const _Float16* bcol0 = Wt + (size_t)(tq * 64 + 0 * 16 + l15) * DMODEL;
    const _Float16* bcol1 = Wt + (size_t)(tq * 64 + 1 * 16 + l15) * DMODEL;
    const _Float16* bcol2 = Wt + (size_t)(tq * 64 + 2 * 16 + l15) * DMODEL;
    const _Float16* bcol3 = Wt + (size_t)(tq * 64 + 3 * 16 + l15) * DMODEL;

    v8f acc[4];
#pragma unroll
    for (int j = 0; j < 4; ++j) acc[j] = zero8();

#pragma unroll 2
    for (int kk = 0; kk < DMODEL; kk += 32) {
        v16h b0 = load16h(bcol0 + kk + hs * 16);
        v16h b1 = load16h(bcol1 + kk + hs * 16);
        v16h b2 = load16h(bcol2 + kk + hs * 16);
        v16h b3 = load16h(bcol3 + kk + hs * 16);
        v16h a  = load_a_f16(&apanel[l15][kk], hs);
        acc[0] = wmma16(a, b0, acc[0]);
        acc[1] = wmma16(a, b1, acc[1]);
        acc[2] = wmma16(a, b2, acc[2]);
        acc[3] = wmma16(a, b3, acc[3]);
    }

#pragma unroll
    for (int j = 0; j < 4; ++j)
#pragma unroll
        for (int v = 0; v < 8; ++v)
            Out[(size_t)(tm * 16 + v + hs * 8) * DMODEL + tq * 64 + j * 16 + l15] =
                acc[j][v];
}

// ---------------------------------------------------------------------------
extern "C" void kernel_launch(void* const* d_in, const int* in_sizes, int n_in,
                              void* d_out, int out_size, void* d_ws, size_t ws_size,
                              hipStream_t stream) {
    const float* q    = (const float*)d_in[0];
    const float* k    = (const float*)d_in[1];
    const float* v    = (const float*)d_in[2];
    const float* mask = (const float*)d_in[3];
    const float* Wq   = (const float*)d_in[4];
    const float* Wk   = (const float*)d_in[5];
    const float* Wv   = (const float*)d_in[6];
    const float* Wo   = (const float*)d_in[7];
    float* out = (float*)d_out;

    const size_t WSZ = (size_t)DMODEL * DMODEL;          // 1M halves per weight
    const size_t TSZ = (size_t)B_SZ * S_LEN * DMODEL;    // 8M halves per tensor
    _Float16* ws = (_Float16*)d_ws;
    _Float16* wt_q = ws;
    _Float16* wt_k = wt_q + WSZ;
    _Float16* wt_v = wt_k + WSZ;
    _Float16* wt_o = wt_v + WSZ;
    _Float16* qh   = wt_o + WSZ;
    _Float16* kh   = qh + TSZ;
    _Float16* vt   = kh + TSZ;
    _Float16* aw   = vt + TSZ;                            // total ~75.5 MB

    // 1) weights -> transposed f16
    transpose_w_kernel<<<4096, 256, 0, stream>>>(Wq, wt_q);
    transpose_w_kernel<<<4096, 256, 0, stream>>>(Wk, wt_k);
    transpose_w_kernel<<<4096, 256, 0, stream>>>(Wv, wt_v);
    transpose_w_kernel<<<4096, 256, 0, stream>>>(Wo, wt_o);

    // 2) projections (1024 blocks x 8 waves; block shares one A panel)
    proj_kernel<<<1024, 256, 0, stream>>>(q, wt_q, qh, 0);
    proj_kernel<<<1024, 256, 0, stream>>>(k, wt_k, kh, 0);
    proj_kernel<<<1024, 256, 0, stream>>>(v, wt_v, vt, 1);

    // 3) flash attention (8192 wave-tiles -> 1024 blocks)
    attn_kernel<<<1024, 256, 0, stream>>>(qh, kh, vt, mask, aw);

    // 4) output projection -> f32
    outproj_kernel<<<1024, 256, 0, stream>>>(aw, wt_o, out);
}
